// BipartiteDataEncoder_19928648254212
// MI455X (gfx1250) — matrix operand order repaired and checked by reading, not compiled, not run on hardware
//
#include <hip/hip_runtime.h>

typedef __attribute__((ext_vector_type(16))) _Float16 v16h;
typedef __attribute__((ext_vector_type(8)))  float    v8f;

#define EMB 64

// ---------------------------------------------------------------- zero fill
__global__ void zero_kernel(float* __restrict__ p, long long n) {
  long long i = (long long)blockIdx.x * blockDim.x + threadIdx.x;
  long long stride = (long long)gridDim.x * blockDim.x;
  for (; i < n; i += stride) p[i] = 0.0f;
}

// ------------------------------------------------- PreNorm + Linear-ReLU x2
// 256 threads = 4 nodes x 64 features. Hidden layer staged through LDS.
template<int NF>
__global__ void embed_mlp_kernel(const float* __restrict__ x,
                                 const float* __restrict__ shift,
                                 const float* __restrict__ scale,
                                 const float* __restrict__ w1,
                                 const float* __restrict__ b1,
                                 const float* __restrict__ w2,
                                 const float* __restrict__ b2,
                                 float* __restrict__ out, int n) {
  __shared__ float hid[4][EMB];
  const int local = threadIdx.x >> 6;
  const int f     = threadIdx.x & 63;
  const int node  = blockIdx.x * 4 + local;
  float h = 0.0f;
  if (node < n) {
    h = b1[f];
#pragma unroll
    for (int i = 0; i < NF; ++i) {
      float xi = (x[(long long)node * NF + i] + shift[i]) * scale[i];
      h = fmaf(xi, w1[i * EMB + f], h);
    }
    h = fmaxf(h, 0.0f);
  }
  hid[local][f] = h;
  __syncthreads();
  if (node < n) {
    float o = b2[f];
#pragma unroll 8
    for (int k = 0; k < EMB; ++k) o = fmaf(hid[local][k], w2[k * EMB + f], o);
    out[(long long)node * EMB + f] = fmaxf(o, 0.0f);
  }
}

// ------------------------------------------------- edge gather + scatter-add
// One thread per (edge, 4-feature chunk): float4 gather (L2-resident source
// table) + 4 f32 atomic adds into the L2-resident segment-sum buffer.
__global__ void sage_aggregate_kernel(const float* __restrict__ x_src,
                                      const int* __restrict__ src_idx,
                                      const int* __restrict__ dst_idx,
                                      float* __restrict__ ssum,
                                      float* cnt,   // nullptr -> skip counting
                                      int n_edges) {
  long long t = (long long)blockIdx.x * blockDim.x + threadIdx.x;
  int e  = (int)(t >> 4);
  int f4 = ((int)t & 15) * 4;
  if (e >= n_edges) return;
  int s = src_idx[e];
  int d = dst_idx[e];
  const float4 v = *(const float4*)(x_src + (long long)s * EMB + f4);
  float* dp = ssum + (long long)d * EMB + f4;
  atomicAdd(dp + 0, v.x);
  atomicAdd(dp + 1, v.y);
  atomicAdd(dp + 2, v.z);
  atomicAdd(dp + 3, v.w);
  if (f4 == 0 && cnt != nullptr) atomicAdd(&cnt[d], 1.0f);
}

// ---------------------------------------- out = relu([mean|x_dst]@[llw;lrw]+b)
// One wave per 16-row tile: A = 16x128 f16 ([ssum/cnt | x_dst]), B = 128x64 f16
// ([ll_w ; lr_w]), 4 K-chunks x 4 N-tiles = 16 v_wmma_f32_16x16x32_f16.
// The shared 128x64 weight block is staged ONCE per workgroup into LDS,
// f16-converted and pre-swizzled into per-lane WMMA B-fragment layout, so
// each fragment fetch is two ds_load_b128. A fragments use float4 loads
// (two contiguous 8-float runs per K-chunk per lane).
// NOTE: ssum/x_dst/out intentionally NOT __restrict__ (in-place update:
// A-fragments are register-resident before any store; tiles are row-disjoint).
__global__ void sage_linear_relu_kernel(const float* ssum,
                                        const float* __restrict__ cnt,
                                        const float* x_dst,
                                        const float* __restrict__ ll_w,
                                        const float* __restrict__ ll_b,
                                        const float* __restrict__ lr_w,
                                        float* out, int n_dst) {
  // 16 fragments (kc 0..3 x nt 0..3) x 32 lanes x 16 f16 = 16 KB
  __shared__ __align__(32) _Float16 wlds[16 * 32 * 16];

  // ---- cooperative weight staging (all 256 threads; before any wave exits)
  {
    unsigned* wlds_u = (unsigned*)wlds;
    const int tid = threadIdx.x;
#pragma unroll
    for (int d = 0; d < 16; ++d) {
      int D  = tid * 16 + d;        // dword index 0..4095
      int jd = D & 7;               // f16-pair index within lane chunk
      int L  = (D >> 3) & 31;       // consuming lane
      int fi = D >> 8;              // fragment index = kc*4 + nt
      int kc = fi >> 2, nt = fi & 3;
      int n  = nt * 16 + (L & 15);
      int K0 = kc * 32 + (L >> 4) * 8 + (jd >> 2) * 8 + jd * 2;   // even; pair never straddles 64
      const float* Wp = (K0 < EMB) ? (ll_w + K0 * EMB + n)
                                   : (lr_w + (K0 - EMB) * EMB + n);
      union { _Float16 h[2]; unsigned u; } pk;
      pk.h[0] = (_Float16)Wp[0];
      pk.h[1] = (_Float16)Wp[EMB];
      wlds_u[D] = pk.u;
    }
  }
  __syncthreads();

  const int wave = threadIdx.x >> 5;
  const int lane = threadIdx.x & 31;
  const int tile = blockIdx.x * 8 + wave;
  const int row0 = tile * 16;
  if (row0 >= n_dst) return;            // wave-uniform; WMMA waves keep EXEC all-1s
  const int hl = lane >> 4;             // lane half
  const int m  = lane & 15;             // A row / B,C column within tile
  const long long rowA = row0 + m;      // n_dst is a multiple of 16 here

  const float invc = 1.0f / fmaxf(cnt[rowA], 1.0f);
  const float* arow_s = ssum  + rowA * EMB;   // K = 0..63  (scaled by invc)
  const float* arow_x = x_dst + rowA * EMB;   // K = 64..127

  // A fragments: per kc, lane needs runs [kb+hl*8, +8) and [kb+16+hl*8, +8)
  v16h afrag[4];
#pragma unroll
  for (int kc = 0; kc < 4; ++kc) {
    const float* base = (kc < 2) ? arow_s : arow_x;
    const float  sc   = (kc < 2) ? invc : 1.0f;
    const int kb = (kc & 1) * 32;
    float va[8], vb[8];
    *(float4*)&va[0] = *(const float4*)(base + kb + hl * 8);
    *(float4*)&va[4] = *(const float4*)(base + kb + hl * 8 + 4);
    *(float4*)&vb[0] = *(const float4*)(base + kb + 16 + hl * 8);
    *(float4*)&vb[4] = *(const float4*)(base + kb + 16 + hl * 8 + 4);
#pragma unroll
    for (int j = 0; j < 8; ++j) {
      afrag[kc][j]     = (_Float16)(va[j] * sc);
      afrag[kc][8 + j] = (_Float16)(vb[j] * sc);
    }
  }

#pragma unroll
  for (int nt = 0; nt < 4; ++nt) {
    const int n = nt * 16 + m;          // global output column for this lane
    const float bias = ll_b[n];
    v8f acc;
#pragma unroll
    for (int r = 0; r < 8; ++r) acc[r] = bias;
#pragma unroll
    for (int kc = 0; kc < 4; ++kc) {
      const v16h bfrag =
          *(const v16h*)&wlds[((kc * 4 + nt) * 32 + lane) * 16];
      acc = __builtin_amdgcn_wmma_f32_16x16x32_f16(
          false, afrag[kc], false, bfrag, (short)0, acc, false, false);
    }
#pragma unroll
    for (int r = 0; r < 8; ++r) {
      long long rowD = row0 + r + 8 * hl;
      out[rowD * EMB + n] = fmaxf(acc[r], 0.0f);
    }
  }
}

// --------------------------------------------------------------------- host
extern "C" void kernel_launch(void* const* d_in, const int* in_sizes, int n_in,
                              void* d_out, int out_size, void* d_ws, size_t ws_size,
                              hipStream_t stream) {
  (void)n_in; (void)out_size; (void)ws_size;
  const float* cons_x     = (const float*)d_in[0];
  const float* var_x      = (const float*)d_in[1];
  const int*   edge_idx   = (const int*)d_in[3];
  const float* cons_shift = (const float*)d_in[4];
  const float* cons_scale = (const float*)d_in[5];
  const float* cons_w1    = (const float*)d_in[6];
  const float* cons_b1    = (const float*)d_in[7];
  const float* cons_w2    = (const float*)d_in[8];
  const float* cons_b2    = (const float*)d_in[9];
  const float* var_shift  = (const float*)d_in[10];
  const float* var_scale  = (const float*)d_in[11];
  const float* var_w1     = (const float*)d_in[12];
  const float* var_b1     = (const float*)d_in[13];
  const float* var_w2     = (const float*)d_in[14];
  const float* var_b2     = (const float*)d_in[15];
  const float* conv_ll_w  = (const float*)d_in[18];   // [2,2,64,64]
  const float* conv_ll_b  = (const float*)d_in[19];   // [2,2,64]
  const float* conv_lr_w  = (const float*)d_in[20];   // [2,2,64,64]

  const int n_cons  = in_sizes[0] / 5;
  const int n_var   = in_sizes[1] / 19;
  const int n_edges = in_sizes[3] / 2;
  const int* src = edge_idx;              // constraint index per edge
  const int* dst = edge_idx + n_edges;    // variable index per edge

  // Workspace carve (256B-aligned): ~103 MB total.
  char* wsp = (char*)d_ws;
  auto carve = [&](size_t bytes) -> float* {
    char* p = wsp;
    wsp += (bytes + 255) & ~(size_t)255;
    return (float*)p;
  };
  float* xc0   = carve((size_t)n_cons * EMB * 4);
  float* xc1   = carve((size_t)n_cons * EMB * 4);
  float* xv0   = carve((size_t)n_var  * EMB * 4);
  float* cnt_c = carve((size_t)n_cons * 4);
  float* cnt_v = carve((size_t)n_var  * 4);
  float* xv1   = (float*)d_out;           // d_out doubles as ssum_v / x_v buffer

  // Embeddings
  embed_mlp_kernel<5><<<(n_cons + 3) / 4, 256, 0, stream>>>(
      cons_x, cons_shift, cons_scale, cons_w1, cons_b1, cons_w2, cons_b2, xc0, n_cons);
  embed_mlp_kernel<19><<<(n_var + 3) / 4, 256, 0, stream>>>(
      var_x, var_shift, var_scale, var_w1, var_b1, var_w2, var_b2, xv0, n_var);

  const long long aggT = (long long)n_edges * 16;
  const int aggBlocks  = (int)((aggT + 255) / 256);
  const int linBlkV    = (n_var / 16 + 7) / 8;
  const int linBlkC    = (n_cons / 16 + 7) / 8;

  // ---------------- layer 0 ----------------
  zero_kernel<<<2048, 256, 0, stream>>>(xv1, (long long)n_var * EMB);
  zero_kernel<<<2048, 256, 0, stream>>>(xc1, (long long)n_cons * EMB);
  zero_kernel<<<256, 256, 0, stream>>>(cnt_v, n_var);
  zero_kernel<<<256, 256, 0, stream>>>(cnt_c, n_cons);
  // cons -> var and var -> cons aggregation (counts computed once; structural)
  sage_aggregate_kernel<<<aggBlocks, 256, 0, stream>>>(xc0, src, dst, xv1, cnt_v, n_edges);
  sage_aggregate_kernel<<<aggBlocks, 256, 0, stream>>>(xv0, dst, src, xc1, cnt_c, n_edges);
  // new_v (in place over ssum_v) and new_c (in place over ssum_c)
  sage_linear_relu_kernel<<<linBlkV, 256, 0, stream>>>(
      xv1, cnt_v, xv0, conv_ll_w + 0 * 4096, conv_ll_b + 0 * 64, conv_lr_w + 0 * 4096,
      xv1, n_var);
  sage_linear_relu_kernel<<<linBlkC, 256, 0, stream>>>(
      xc1, cnt_c, xc0, conv_ll_w + 1 * 4096, conv_ll_b + 1 * 64, conv_lr_w + 1 * 4096,
      xc1, n_cons);

  // ---------------- layer 1 (only x_v is returned; cons update is dead) ----
  zero_kernel<<<2048, 256, 0, stream>>>(xv0, (long long)n_var * EMB);
  sage_aggregate_kernel<<<aggBlocks, 256, 0, stream>>>(xc1, src, dst, xv0,
                                                       nullptr, n_edges);
  sage_linear_relu_kernel<<<linBlkV, 256, 0, stream>>>(
      xv0, cnt_v, xv1, conv_ll_w + 2 * 4096, conv_ll_b + 2 * 64, conv_lr_w + 2 * 4096,
      (float*)d_out, n_var);
}
